// AttentionBlock_14388140442071
// MI455X (gfx1250) — matrix operand order repaired and checked by reading, not compiled, not run on hardware
//
#include <hip/hip_runtime.h>

typedef __attribute__((ext_vector_type(16))) _Float16 v16h;
typedef __attribute__((ext_vector_type(8)))  _Float16 v8h;
typedef __attribute__((ext_vector_type(8)))  float    v8f;
typedef __attribute__((ext_vector_type(4)))  float    v4f;
typedef __attribute__((ext_vector_type(4)))  int      i32x4;

typedef __attribute__((address_space(1))) i32x4 gi32x4;   // global int4
typedef __attribute__((address_space(3))) i32x4 li32x4;   // LDS int4

#define B_ 8
#define C_ 256
#define N_ 4096
#define G_ 8

#if defined(__has_builtin)
#if __has_builtin(__builtin_amdgcn_global_load_async_to_lds_b128)
#define HAVE_ASYNC_LDS 1
#endif
#if __has_builtin(__builtin_amdgcn_s_wait_asynccnt)
#define HAVE_WAIT_ASYNC 1
#endif
#endif

static __device__ __forceinline__ v8f wmma16(v16h a, v16h b, v8f c) {
  // D = A x B + C, f16 inputs, f32 accumulate (v_wmma_f32_16x16x32_f16)
  return __builtin_amdgcn_wmma_f32_16x16x32_f16(false, a, false, b, (short)0, c, false, false);
}

// A-fragment (16x32 f16), rows K-major with leading dim ld (elements).
// Lane layout: row = lane&15, half h=lane>>4 holds K = 16h..16h+15 (contiguous, matches ldB).
static __device__ __forceinline__ v16h ldA(const _Float16* p, int ld) {
  int lane = threadIdx.x & 31;
  const _Float16* r = p + (size_t)(lane & 15) * ld + (lane >> 4) * 16;
  v8h lo = *(const v8h*)r;
  v8h hi = *(const v8h*)(r + 8);
  v16h f;
#pragma unroll
  for (int i = 0; i < 8; ++i) { f[i] = lo[i]; f[i + 8] = hi[i]; }
  return f;
}

// B-fragment (32x16 f16) loaded from Bt stored N-major over K (row n holds K 0..31).
// Lane layout: col = lane&15, half h=lane>>4 holds K = 16h..16h+15 (one contiguous 32B run).
static __device__ __forceinline__ v16h ldB(const _Float16* p, int ld) {
  int lane = threadIdx.x & 31;
  const _Float16* r = p + (size_t)(lane & 15) * ld + (lane >> 4) * 16;
  return *(const v16h*)r;
}

// memory -> LDS direct copy, 16B per lane. Uses CDNA5 GLOBAL_LOAD_ASYNC_TO_LDS_B128
// (tracked by ASYNCcnt) when the builtin is available; else VGPR round-trip fallback.
static __device__ __forceinline__ void g2l_b128(const _Float16* g, _Float16* l) {
#if defined(HAVE_ASYNC_LDS)
  __builtin_amdgcn_global_load_async_to_lds_b128((gi32x4*)g, (li32x4*)l, 0, 0);
#else
  *(v8h*)l = *(const v8h*)g;
#endif
}

static __device__ __forceinline__ void drain_g2l() {
#if defined(HAVE_ASYNC_LDS)
#if defined(HAVE_WAIT_ASYNC)
  __builtin_amdgcn_s_wait_asynccnt(0);
#else
  asm volatile("s_wait_asynccnt 0" ::: "memory");
#endif
#endif
}

// ---------------------------------------------------------------------------
// K0: weights f32 -> f16 (layouts are already (o, c) row-major == K-major Bt)
// ---------------------------------------------------------------------------
__global__ void cvt_w_kernel(const float* __restrict__ qkvw, const float* __restrict__ projw,
                             _Float16* __restrict__ wq, _Float16* __restrict__ wp) {
  int i = blockIdx.x * blockDim.x + threadIdx.x;
  if (i < 3 * C_ * C_) wq[i] = (_Float16)qkvw[i];
  if (i < C_ * C_)     wp[i] = (_Float16)projw[i];
}

// ---------------------------------------------------------------------------
// K1: GroupNorm(8) -> xn f16, token-major (b, n, c)
// ---------------------------------------------------------------------------
__global__ void groupnorm_kernel(const float* __restrict__ x, const float* __restrict__ gsc,
                                 const float* __restrict__ gbi, _Float16* __restrict__ xn) {
  const int b = blockIdx.x >> 3, g = blockIdx.x & 7;
  const int CPG = C_ / G_;              // 32 channels per group
  const int CNT = CPG * N_;             // 131072 elements per (b, g)
  const float* xg = x + ((size_t)b * C_ + (size_t)g * CPG) * N_;

  float s = 0.f, s2 = 0.f;
  const v4f* p4 = (const v4f*)xg;
  for (int i = threadIdx.x; i < CNT / 4; i += blockDim.x) {
    v4f v = p4[i];
    s  += v.x + v.y + v.z + v.w;
    s2 += v.x * v.x + v.y * v.y + v.z * v.z + v.w * v.w;
  }
  __shared__ float rs[256], rq[256];
  rs[threadIdx.x] = s; rq[threadIdx.x] = s2;
  __syncthreads();
  for (int st = 128; st > 0; st >>= 1) {
    if ((int)threadIdx.x < st) {
      rs[threadIdx.x] += rs[threadIdx.x + st];
      rq[threadIdx.x] += rq[threadIdx.x + st];
    }
    __syncthreads();
  }
  __shared__ float sm, sr;
  if (threadIdx.x == 0) {
    float mean = rs[0] / (float)CNT;
    float var  = rq[0] / (float)CNT - mean * mean;
    sm = mean; sr = rsqrtf(var + 1e-5f);
  }
  __syncthreads();
  float mean = sm, rinv = sr;
  // write (b, n, c) f16: c fastest across threads -> contiguous 2B*32 runs
  for (int i = threadIdx.x; i < CNT; i += blockDim.x) {
    int cc = i & (CPG - 1), sp = i >> 5;
    int c  = g * CPG + cc;
    float v = xg[(size_t)cc * N_ + sp];
    float y = (v - mean) * rinv * gsc[c] + gbi[c];
    xn[((size_t)b * N_ + sp) * C_ + c] = (_Float16)y;
  }
}

// ---------------------------------------------------------------------------
// K2: QKV GEMM  (n x c) * (3c x c)^T + bias.  32x32 tile per wave.
// Writes: q (b,n,c) f16 pre-scaled by 1/sqrt(C); k (b,n,c) f16; v transposed (b,c,n) f16.
// ---------------------------------------------------------------------------
__global__ void qkv_gemm_kernel(const _Float16* __restrict__ xn, const _Float16* __restrict__ w,
                                const float* __restrict__ bias,
                                _Float16* __restrict__ q, _Float16* __restrict__ k,
                                _Float16* __restrict__ vT) {
  const int NT = (3 * C_) / 32;         // 24
  const int MT = N_ / 32;               // 128
  int wave = threadIdx.x >> 5, lane = threadIdx.x & 31;
  int tile = blockIdx.x * (blockDim.x >> 5) + wave;
  int b = tile / (MT * NT); tile -= b * MT * NT;
  int tm = tile / NT, tn = tile - tm * NT;
  int m0 = tm * 32, n0 = tn * 32;
  const _Float16* A  = xn + ((size_t)b * N_ + m0) * C_;
  const _Float16* Bt = w + (size_t)n0 * C_;

  v8f acc[2][2] = {};
#pragma unroll 4
  for (int kk = 0; kk < C_; kk += 32) {
    __builtin_prefetch(A + kk + 128, 0, 0);
    v16h a0 = ldA(A + kk, C_);
    v16h a1 = ldA(A + 16 * C_ + kk, C_);
    v16h b0 = ldB(Bt + kk, C_);
    v16h b1 = ldB(Bt + 16 * C_ + kk, C_);
    acc[0][0] = wmma16(a0, b0, acc[0][0]);
    acc[0][1] = wmma16(a0, b1, acc[0][1]);
    acc[1][0] = wmma16(a1, b0, acc[1][0]);
    acc[1][1] = wmma16(a1, b1, acc[1][1]);
  }
  int col = lane & 15, h = lane >> 4;
#pragma unroll
  for (int mi = 0; mi < 2; ++mi) {
#pragma unroll
    for (int ni = 0; ni < 2; ++ni) {
      int oc = n0 + ni * 16 + col;
      float bs = bias[oc];
      int tb = m0 + mi * 16 + 8 * h;        // 8 contiguous token rows per lane
      if (oc < C_) {                        // Q: fold softmax scale C^-0.5 = 1/16
#pragma unroll
        for (int r = 0; r < 8; ++r)
          q[((size_t)b * N_ + tb + r) * C_ + oc] = (_Float16)((acc[mi][ni][r] + bs) * 0.0625f);
      } else if (oc < 2 * C_) {
        int ock = oc - C_;
#pragma unroll
        for (int r = 0; r < 8; ++r)
          k[((size_t)b * N_ + tb + r) * C_ + ock] = (_Float16)(acc[mi][ni][r] + bs);
      } else {                              // V transposed (b, c, n): contiguous b128 store
        int cv = oc - 2 * C_;
        v8h pk;
#pragma unroll
        for (int r = 0; r < 8; ++r) pk[r] = (_Float16)(acc[mi][ni][r] + bs);
        *(v8h*)(vT + ((size_t)b * C_ + cv) * N_ + tb) = pk;
      }
    }
  }
}

// ---------------------------------------------------------------------------
// K3: fused flash attention. WG = 8 waves x 16 query rows = 128 tokens.
// Per 32-key chunk: async-stage K/V to LDS, S = Q K^T (WMMA), online softmax
// (shfl_xor butterflies), P relayout D->A via per-wave LDS, O += P V^T (WMMA).
// ---------------------------------------------------------------------------
__global__ void __launch_bounds__(256) attn_kernel(const _Float16* __restrict__ q,
                                                   const _Float16* __restrict__ k,
                                                   const _Float16* __restrict__ vT,
                                                   _Float16* __restrict__ ao) {
  __shared__ __align__(32) _Float16 sK[32 * C_];       // 16 KB: 32 keys x 256 ch (K-major rows)
  __shared__ __align__(32) _Float16 sV[C_ * 32];       // 16 KB: 256 ch x 32 keys (Vt rows)
  __shared__ __align__(32) _Float16 sP[8 * 16 * 32];   //  8 KB: per-wave P staging

  int b = blockIdx.x >> 5, mblk = blockIdx.x & 31;
  int wave = threadIdx.x >> 5, lane = threadIdx.x & 31;
  int col = lane & 15, h = lane >> 4;

  // Q fragments resident in VGPRs (16 rows x 256 K = 8 A-frags)
  const _Float16* Qw = q + ((size_t)b * N_ + mblk * 128 + wave * 16) * C_;
  v16h qa[8];
#pragma unroll
  for (int i = 0; i < 8; ++i) qa[i] = ldA(Qw + i * 32, C_);

  const _Float16* Kb = k + (size_t)b * N_ * C_;
  const _Float16* Vb = vT + (size_t)b * C_ * N_;
  _Float16* sPw = sP + wave * 16 * 32;

  v8f o[16] = {};
  float mrun[8], lrun[8];
#pragma unroll
  for (int r = 0; r < 8; ++r) { mrun[r] = -1e30f; lrun[r] = 0.f; }

  for (int j = 0; j < N_; j += 32) {
    __syncthreads();  // all waves done reading previous chunk
    {
      // K chunk is fully contiguous (32 token rows x 256 ch); V^T chunk is 256
      // rows of 32 f16 at stride N_. 16B per lane per issue, ASYNCcnt-tracked.
      const _Float16* srcK = Kb + (size_t)j * C_;
      for (int i = threadIdx.x; i < 1024; i += 256)
        g2l_b128(srcK + (size_t)i * 8, sK + (size_t)i * 8);
      for (int i = threadIdx.x; i < 1024; i += 256) {
        int c = i >> 2, part = i & 3;
        g2l_b128(Vb + (size_t)c * N_ + j + part * 8, sV + (size_t)i * 8);
      }
      drain_g2l();  // s_wait_asynccnt 0 (per wave) before publishing via barrier
    }
    __syncthreads();

    // S = Q x Kchunk^T : 16 x 32 scores (two 16x16 D-frags)
    v8f s0 = {}, s1 = {};
#pragma unroll
    for (int kk = 0; kk < 8; ++kk) {
      v16h b0 = ldB(sK + kk * 32, C_);
      v16h b1 = ldB(sK + 16 * C_ + kk * 32, C_);
      s0 = wmma16(qa[kk], b0, s0);
      s1 = wmma16(qa[kk], b1, s1);
    }

    // online softmax: per-lane r indexes row M = r + 8h of this wave's 16 rows
    float cmax[8], alpha[8], csum[8];
#pragma unroll
    for (int r = 0; r < 8; ++r) cmax[r] = fmaxf(s0[r], s1[r]);
#pragma unroll
    for (int m = 1; m <= 8; m <<= 1)
#pragma unroll
      for (int r = 0; r < 8; ++r) cmax[r] = fmaxf(cmax[r], __shfl_xor(cmax[r], m));
#pragma unroll
    for (int r = 0; r < 8; ++r) {
      float mn = fmaxf(mrun[r], cmax[r]);
      alpha[r] = __expf(mrun[r] - mn);
      mrun[r] = mn;
    }
#pragma unroll
    for (int r = 0; r < 8; ++r) {
      s0[r] = __expf(s0[r] - mrun[r]);
      s1[r] = __expf(s1[r] - mrun[r]);
      csum[r] = s0[r] + s1[r];
    }
#pragma unroll
    for (int m = 1; m <= 8; m <<= 1)
#pragma unroll
      for (int r = 0; r < 8; ++r) csum[r] += __shfl_xor(csum[r], m);
#pragma unroll
    for (int r = 0; r < 8; ++r) lrun[r] = lrun[r] * alpha[r] + csum[r];
#pragma unroll
    for (int f = 0; f < 16; ++f)
#pragma unroll
      for (int r = 0; r < 8; ++r) o[f][r] *= alpha[r];

    // relayout P: D-fragment (col per lane) -> A-fragment (row per lane) via per-wave LDS
#pragma unroll
    for (int r = 0; r < 8; ++r) {
      int row = r + 8 * h;
      sPw[row * 32 + col]      = (_Float16)s0[r];
      sPw[row * 32 + col + 16] = (_Float16)s1[r];
    }
    asm volatile("s_wait_dscnt 0" ::: "memory");  // cross-lane LDS RAW within wave
    v16h pa = ldA(sPw, 32);

    // O += P x Vt : 16 x 256 in 16 N'-frags, single K=32 step
#pragma unroll
    for (int f = 0; f < 16; ++f) {
      v16h vb = ldB(sV + f * 16 * 32, 32);
      o[f] = wmma16(pa, vb, o[f]);
    }
  }

  // epilogue: O /= l, write (b, n, c) f16 (K-major for proj GEMM)
  float inv[8];
#pragma unroll
  for (int r = 0; r < 8; ++r) inv[r] = 1.f / lrun[r];
  size_t obase = ((size_t)b * N_ + mblk * 128 + wave * 16) * C_;
#pragma unroll
  for (int f = 0; f < 16; ++f) {
    int c = f * 16 + col;
#pragma unroll
    for (int r = 0; r < 8; ++r) {
      int t = r + 8 * h;
      ao[obase + (size_t)t * C_ + c] = (_Float16)(o[f][r] * inv[r]);
    }
  }
}

// ---------------------------------------------------------------------------
// K4: proj GEMM + bias + residual, f32 output in (b, c, h*w) layout.
// D-fragment rows are contiguous tokens -> aligned 32B residual load + store.
// ---------------------------------------------------------------------------
__global__ void proj_gemm_kernel(const _Float16* __restrict__ ao, const _Float16* __restrict__ w,
                                 const float* __restrict__ bias, const float* __restrict__ x,
                                 float* __restrict__ out) {
  const int NT = C_ / 32;               // 8
  const int MT = N_ / 32;               // 128
  int wave = threadIdx.x >> 5, lane = threadIdx.x & 31;
  int tile = blockIdx.x * (blockDim.x >> 5) + wave;
  int b = tile / (MT * NT); tile -= b * MT * NT;
  int tm = tile / NT, tn = tile - tm * NT;
  int m0 = tm * 32, n0 = tn * 32;
  const _Float16* A  = ao + ((size_t)b * N_ + m0) * C_;
  const _Float16* Bt = w + (size_t)n0 * C_;

  v8f acc[2][2] = {};
#pragma unroll 4
  for (int kk = 0; kk < C_; kk += 32) {
    __builtin_prefetch(A + kk + 128, 0, 0);
    v16h a0 = ldA(A + kk, C_);
    v16h a1 = ldA(A + 16 * C_ + kk, C_);
    v16h b0 = ldB(Bt + kk, C_);
    v16h b1 = ldB(Bt + 16 * C_ + kk, C_);
    acc[0][0] = wmma16(a0, b0, acc[0][0]);
    acc[0][1] = wmma16(a0, b1, acc[0][1]);
    acc[1][0] = wmma16(a1, b0, acc[1][0]);
    acc[1][1] = wmma16(a1, b1, acc[1][1]);
  }
  int col = lane & 15, h = lane >> 4;
#pragma unroll
  for (int mi = 0; mi < 2; ++mi) {
#pragma unroll
    for (int ni = 0; ni < 2; ++ni) {
      int oc = n0 + ni * 16 + col;
      float bs = bias[oc];
      int tb = m0 + mi * 16 + 8 * h;
      size_t base = ((size_t)b * C_ + oc) * N_ + tb;
      v8f xr = *(const v8f*)(x + base);
      v8f res;
#pragma unroll
      for (int r = 0; r < 8; ++r) res[r] = acc[mi][ni][r] + bs + xr[r];
      *(v8f*)(out + base) = res;
    }
  }
}

// ---------------------------------------------------------------------------
extern "C" void kernel_launch(void* const* d_in, const int* in_sizes, int n_in,
                              void* d_out, int out_size, void* d_ws, size_t ws_size,
                              hipStream_t stream) {
  const float* x    = (const float*)d_in[0];
  const float* gsc  = (const float*)d_in[1];
  const float* gbi  = (const float*)d_in[2];
  const float* qkvw = (const float*)d_in[3];
  const float* qkvb = (const float*)d_in[4];
  const float* pw   = (const float*)d_in[5];
  const float* pb   = (const float*)d_in[6];
  float* out = (float*)d_out;

  // scratch layout (f16), ~84.4 MB total
  char* ws = (char*)d_ws;
  _Float16* wq  = (_Float16*)ws;                               // 3C*C
  _Float16* wp  = (_Float16*)(ws + (size_t)3 * C_ * C_ * 2);   // C*C
  _Float16* xn  = (_Float16*)(ws + (size_t)4 * C_ * C_ * 2);   // B*N*C
  size_t big = (size_t)B_ * N_ * C_;
  _Float16* q16 = xn + big;
  _Float16* k16 = q16 + big;
  _Float16* vT  = k16 + big;
  _Float16* aot = vT + big;

  cvt_w_kernel<<<(3 * C_ * C_ + 255) / 256, 256, 0, stream>>>(qkvw, pw, wq, wp);
  groupnorm_kernel<<<B_ * G_, 256, 0, stream>>>(x, gsc, gbi, xn);
  qkv_gemm_kernel<<<(B_ * (N_ / 32) * (3 * C_ / 32)) / 8, 256, 0, stream>>>(xn, wq, qkvb, q16, k16, vT);
  attn_kernel<<<B_ * (N_ / 128), 256, 0, stream>>>(q16, k16, vT, aot);
  proj_gemm_kernel<<<(B_ * (N_ / 32) * (C_ / 32)) / 8, 256, 0, stream>>>(aot, wp, pb, x, out);
}
